// CoconutBinaryLatentModel_81286551044480
// MI455X (gfx1250) — matrix operand order repaired and checked by reading, not compiled, not run on hardware
//
#include <hip/hip_runtime.h>
#include <hip/hip_bf16.h>
#include <math.h>

// ---------------------------------------------------------------------------
// Types for CDNA5 WMMA (wave32): v_wmma_f32_16x16x32_bf16
// ---------------------------------------------------------------------------
typedef __bf16 bf16;
typedef __attribute__((ext_vector_type(16))) __bf16 bf16x16;
typedef __attribute__((ext_vector_type(8)))  __bf16 bf16x8;
typedef __attribute__((ext_vector_type(8)))  float  f32x8;
typedef unsigned int u32x4 __attribute__((ext_vector_type(4)));
typedef int          i32x4 __attribute__((ext_vector_type(4)));
typedef int          i32x8 __attribute__((ext_vector_type(8)));

constexpr int TBM = 128;   // block tile M
constexpr int TBN = 128;   // block tile N
constexpr int TBK = 32;    // k step (== WMMA K)

#if defined(__HIP_DEVICE_COMPILE__) && __has_builtin(__builtin_amdgcn_tensor_load_to_lds) && __has_builtin(__builtin_amdgcn_s_wait_tensorcnt)
#define HAVE_TDM 1
#else
#define HAVE_TDM 0
#endif

// ---------------------------------------------------------------------------
// Weight-GEMM: C = act(A @ Wt^T (+bias)) (+resid)
//   A  : fp32 [M,K] (lda == K), loaded to registers then converted to bf16
//   Wt : bf16 [N,K] (pre-transposed weights) -> B tile DMA'd by the TDM
// Double-buffered pipeline: TDM(t+1) + A-loads(t+1) overlap WMMA burst of t.
// All M,N,K are exact multiples of the tile, so loads/stores are unguarded.
// ---------------------------------------------------------------------------
__global__ __launch_bounds__(256)
void gemm_wmma_wbf16_kernel(const float* __restrict__ A, const bf16* __restrict__ Wt,
                            float* __restrict__ C, const float* __restrict__ bias,
                            const float* __restrict__ resid,
                            int M, int N, int K, int relu)
{
    __shared__ bf16 As[2][TBM * TBK];   // [m][k]
    __shared__ bf16 Bs[2][TBN * TBK];   // [n][k]  (n-major, matches TDM row walk)

    const int blockM = blockIdx.y * TBM;
    const int blockN = blockIdx.x * TBN;
    const int tid  = threadIdx.x;
    const int w    = tid >> 5;
    const int lane = tid & 31;
    const int wm   = w & 1;
    const int wn   = w >> 1;
    const int g    = lane >> 4;
    const int l16  = lane & 15;

    f32x8 acc[4][2];
    #pragma unroll
    for (int mi = 0; mi < 4; mi++)
        #pragma unroll
        for (int ni = 0; ni < 2; ni++)
            #pragma unroll
            for (int r = 0; r < 8; r++)
                acc[mi][ni][r] = 0.0f;

#if HAVE_TDM
    // LDS aperture keeps the byte offset in addr[31:0] (ISA 10.2), so a
    // truncated generic pointer to the shared array is the D# lds_addr.
    const unsigned ldsB0 = (unsigned)(size_t)(void*)&Bs[0][0];
    const unsigned ldsB1 = (unsigned)(size_t)(void*)&Bs[1][0];
#endif

    float4 areg[4];

    auto loadA = [&](int k0) {
        #pragma unroll
        for (int j = 0; j < 4; j++) {
            int q   = tid + j * 256;
            int row = q >> 3;
            int col = (q & 7) << 2;
            areg[j] = *(const float4*)(A + (size_t)(blockM + row) * K + k0 + col);
        }
    };
    auto storeA = [&](int buf) {
        #pragma unroll
        for (int j = 0; j < 4; j++) {
            int q   = tid + j * 256;
            int row = q >> 3;
            int col = (q & 7) << 2;
            bf16* dst = &As[buf][row * TBK + col];
            dst[0] = (bf16)areg[j].x; dst[1] = (bf16)areg[j].y;
            dst[2] = (bf16)areg[j].z; dst[3] = (bf16)areg[j].w;
        }
    };
    auto loadB = [&](int buf, int k0) {
#if HAVE_TDM
        if (tid < 32) {  // wave 0 issues the tensor DMA for the whole B tile
            unsigned long long ga =
                (unsigned long long)(size_t)(Wt + (size_t)blockN * K + k0);
            u32x4 g0;
            g0[0] = 1u;                                   // count=1, no gather
            g0[1] = buf ? ldsB1 : ldsB0;                  // lds_addr
            g0[2] = (unsigned)ga;                         // global_addr[31:0]
            g0[3] = (unsigned)((ga >> 32) & 0x01FFFFFFull) | (2u << 30); // hi + type=2
            i32x8 g1;
            g1[0] = 0x00010000;          // workgroup_mask=0, data_size=1 (2 bytes)
            g1[1] = 0;                   // abar=0, tensor_dim0 lo16 (td0 = 1<<30)
            g1[2] = 0x4000;              // td0 hi16 = 0x4000, td1 lo16 = 0
            g1[3] = 0x4000 | (TBK << 16);// td1 hi16 (1<<30), tile_dim0 = 32
            g1[4] = TBN;                 // tile_dim1 = 128, tile_dim2 = 0
            g1[5] = K;                   // tensor_dim0_stride = K elements
            g1[6] = 0;
            g1[7] = 0;
            i32x4 z4 = {0, 0, 0, 0};
#if __clang_major__ >= 23
            i32x8 z8 = {0, 0, 0, 0, 0, 0, 0, 0};
            __builtin_amdgcn_tensor_load_to_lds(g0, g1, z4, z4, z8, 0);
#else
            __builtin_amdgcn_tensor_load_to_lds(g0, g1, z4, z4, 0);
#endif
        }
#else
        #pragma unroll
        for (int j = 0; j < 2; j++) {
            int c  = tid + j * 256;           // 512 chunks of 8 bf16
            int n  = c >> 2;
            int kc = (c & 3) << 3;
            *(bf16x8*)&Bs[buf][n * TBK + kc] =
                *(const bf16x8*)(Wt + (size_t)(blockN + n) * K + k0 + kc);
        }
#endif
    };

    // ---- prologue: stage tile 0 into buffer 0 ----
    loadB(0, 0);
    loadA(0);
    storeA(0);
#if HAVE_TDM
    if (tid < 32) __builtin_amdgcn_s_wait_tensorcnt(0);
#endif
    __syncthreads();

    const int numT = K / TBK;
    for (int t = 0; t < numT; t++) {
        const int c  = t & 1;
        const int nx = c ^ 1;
        const bool hasNext = (t + 1 < numT);

        // stage tile t+1 while tile t computes
        if (hasNext) {
            loadB(nx, (t + 1) * TBK);
            loadA((t + 1) * TBK);
        }

        // ---- fragments: A and B both use k = 16*g + s (contiguous) ----
        bf16x16 afrag[4];
        #pragma unroll
        for (int mi = 0; mi < 4; mi++) {
            int m = wm * 64 + mi * 16 + l16;
            union { bf16x16 v; bf16x8 h[2]; } u;
            u.h[0] = *(const bf16x8*)&As[c][m * TBK + 16 * g];
            u.h[1] = *(const bf16x8*)&As[c][m * TBK + 16 * g + 8];
            afrag[mi] = u.v;
        }
        bf16x16 bfrag[2];
        #pragma unroll
        for (int ni = 0; ni < 2; ni++) {
            int n = wn * 32 + ni * 16 + l16;
            bfrag[ni] = *(const bf16x16*)&Bs[c][n * TBK + 16 * g];
        }
        #pragma unroll
        for (int mi = 0; mi < 4; mi++)
            #pragma unroll
            for (int ni = 0; ni < 2; ni++)
                acc[mi][ni] = __builtin_amdgcn_wmma_f32_16x16x32_bf16(
                    false, afrag[mi], false, bfrag[ni],
                    (short)0, acc[mi][ni], false, false);

        if (hasNext) {
            storeA(nx);
#if HAVE_TDM
            if (tid < 32) __builtin_amdgcn_s_wait_tensorcnt(0);
#endif
        }
        __syncthreads();
    }

    // ---- epilogue (exact tiles, no guards) ----
    #pragma unroll
    for (int mi = 0; mi < 4; mi++) {
        #pragma unroll
        for (int ni = 0; ni < 2; ni++) {
            int col = blockN + wn * 32 + ni * 16 + l16;
            #pragma unroll
            for (int r = 0; r < 8; r++) {
                int row = blockM + wm * 64 + mi * 16 + r + 8 * g;
                float v = acc[mi][ni][r];
                if (bias)  v += bias[col];
                if (relu)  v = fmaxf(v, 0.f);
                if (resid) v += resid[(size_t)row * N + col];
                C[(size_t)row * N + col] = v;
            }
        }
    }
}

// ---------------------------------------------------------------------------
// Batched GEMM for attention (both operands fp32 activations):
// C = A @ B.  B: [K,N] ldb (transB=0) or [N,K] ldb (transB=1).
// Batch z: off = (z/inner)*Outer + (z%inner)*Inner per operand.
// Loads are split into load-phase (registers) + convert/store-phase so all
// four global loads stay in flight.
// ---------------------------------------------------------------------------
__global__ __launch_bounds__(256)
void gemm_wmma_kernel(const float* __restrict__ A, const float* __restrict__ Bm,
                      float* __restrict__ C,
                      int M, int N, int K, int lda, int ldb, int ldc,
                      int transB,
                      long long aO, long long aI, long long bO, long long bI,
                      long long cO, long long cI, int inner)
{
    __shared__ bf16 As[TBM * TBK];
    __shared__ bf16 Bs[TBN * TBK];

    int z  = blockIdx.z;
    int zo = z / inner, zi = z % inner;
    A  += (long long)zo * aO + (long long)zi * aI;
    Bm += (long long)zo * bO + (long long)zi * bI;
    C  += (long long)zo * cO + (long long)zi * cI;

    const int blockM = blockIdx.y * TBM;
    const int blockN = blockIdx.x * TBN;
    const int tid  = threadIdx.x;
    const int w    = tid >> 5;
    const int lane = tid & 31;
    const int wm   = w & 1;
    const int wn   = w >> 1;
    const int g    = lane >> 4;
    const int l16  = lane & 15;

    f32x8 acc[4][2];
    #pragma unroll
    for (int mi = 0; mi < 4; mi++)
        #pragma unroll
        for (int ni = 0; ni < 2; ni++)
            #pragma unroll
            for (int r = 0; r < 8; r++)
                acc[mi][ni][r] = 0.0f;

    for (int k0 = 0; k0 < K; k0 += TBK) {
        float4 av[4], bv[4];
        // ---- load phase ----
        #pragma unroll
        for (int j = 0; j < 4; j++) {
            int q   = tid + j * 256;
            int row = q >> 3;
            int col = (q & 7) << 2;
            int gr = blockM + row, gc = k0 + col;
            av[j] = make_float4(0.f, 0.f, 0.f, 0.f);
            if (gr < M && gc < K) av[j] = *(const float4*)(A + (size_t)gr * lda + gc);
        }
        if (!transB) {
            #pragma unroll
            for (int j = 0; j < 4; j++) {
                int q = tid + j * 256;
                int k = q >> 5;
                int n = (q & 31) << 2;
                int gk = k0 + k, gn = blockN + n;
                bv[j] = make_float4(0.f, 0.f, 0.f, 0.f);
                if (gn < N && gk < K) bv[j] = *(const float4*)(Bm + (size_t)gk * ldb + gn);
            }
        } else {
            #pragma unroll
            for (int j = 0; j < 4; j++) {
                int q = tid + j * 256;
                int n = q >> 3;
                int k = (q & 7) << 2;
                int gn = blockN + n, gk = k0 + k;
                bv[j] = make_float4(0.f, 0.f, 0.f, 0.f);
                if (gn < N && gk < K) bv[j] = *(const float4*)(Bm + (size_t)gn * ldb + gk);
            }
        }
        // ---- convert/store phase ----
        #pragma unroll
        for (int j = 0; j < 4; j++) {
            int q   = tid + j * 256;
            int row = q >> 3;
            int col = (q & 7) << 2;
            bf16* dst = &As[row * TBK + col];
            dst[0] = (bf16)av[j].x; dst[1] = (bf16)av[j].y;
            dst[2] = (bf16)av[j].z; dst[3] = (bf16)av[j].w;
        }
        if (!transB) {
            #pragma unroll
            for (int j = 0; j < 4; j++) {
                int q = tid + j * 256;
                int k = q >> 5;
                int n = (q & 31) << 2;
                Bs[(n + 0) * TBK + k] = (bf16)bv[j].x;
                Bs[(n + 1) * TBK + k] = (bf16)bv[j].y;
                Bs[(n + 2) * TBK + k] = (bf16)bv[j].z;
                Bs[(n + 3) * TBK + k] = (bf16)bv[j].w;
            }
        } else {
            #pragma unroll
            for (int j = 0; j < 4; j++) {
                int q = tid + j * 256;
                int n = q >> 3;
                int k = (q & 7) << 2;
                bf16* dst = &Bs[n * TBK + k];
                dst[0] = (bf16)bv[j].x; dst[1] = (bf16)bv[j].y;
                dst[2] = (bf16)bv[j].z; dst[3] = (bf16)bv[j].w;
            }
        }
        __syncthreads();

        bf16x16 afrag[4];
        #pragma unroll
        for (int mi = 0; mi < 4; mi++) {
            int m = wm * 64 + mi * 16 + l16;
            union { bf16x16 v; bf16x8 h[2]; } u;
            u.h[0] = *(const bf16x8*)&As[m * TBK + 16 * g];
            u.h[1] = *(const bf16x8*)&As[m * TBK + 16 * g + 8];
            afrag[mi] = u.v;
        }
        bf16x16 bfrag[2];
        #pragma unroll
        for (int ni = 0; ni < 2; ni++) {
            int n = wn * 32 + ni * 16 + l16;
            bfrag[ni] = *(const bf16x16*)&Bs[n * TBK + 16 * g];
        }
        #pragma unroll
        for (int mi = 0; mi < 4; mi++)
            #pragma unroll
            for (int ni = 0; ni < 2; ni++)
                acc[mi][ni] = __builtin_amdgcn_wmma_f32_16x16x32_bf16(
                    false, afrag[mi], false, bfrag[ni],
                    (short)0, acc[mi][ni], false, false);

        __syncthreads();
    }

    #pragma unroll
    for (int mi = 0; mi < 4; mi++) {
        #pragma unroll
        for (int ni = 0; ni < 2; ni++) {
            int col = blockN + wn * 32 + ni * 16 + l16;
            #pragma unroll
            for (int r = 0; r < 8; r++) {
                int row = blockM + wm * 64 + mi * 16 + r + 8 * g;
                if (row < M && col < N)
                    C[(size_t)row * ldc + col] = acc[mi][ni][r];
            }
        }
    }
}

// ---------------------------------------------------------------------------
// fp32 [K,N] -> bf16 [N,K] transpose-convert (LDS-staged, 32x32 tiles)
// ---------------------------------------------------------------------------
__global__ __launch_bounds__(256)
void convtr_kernel(const float* __restrict__ in, bf16* __restrict__ out, int K, int N)
{
    __shared__ float t[32][33];
    int kb = blockIdx.y * 32, nb = blockIdx.x * 32;
    int tx = threadIdx.x & 31, ty = threadIdx.x >> 5;   // 32 x 8
    #pragma unroll
    for (int i = 0; i < 32; i += 8) {
        int k = kb + ty + i, n = nb + tx;
        t[ty + i][tx] = (k < K && n < N) ? in[(size_t)k * N + n] : 0.f;
    }
    __syncthreads();
    #pragma unroll
    for (int i = 0; i < 32; i += 8) {
        int n = nb + ty + i, k = kb + tx;
        if (n < N && k < K) out[(size_t)n * K + k] = (bf16)t[tx][ty + i];
    }
}

// ---------------------------------------------------------------------------
// Boundary head: prob = sigmoid(h . W2 + b2); hard = prob > 0.5
// ---------------------------------------------------------------------------
__global__ __launch_bounds__(256)
void boundary_kernel(const float* __restrict__ H, const float* __restrict__ W2,
                     const float* __restrict__ b2, int* __restrict__ hb, int HP)
{
    int t    = blockIdx.x * 8 + (threadIdx.x >> 5);
    int lane = threadIdx.x & 31;
    const float* h = H + (size_t)t * HP;
    float s = 0.f;
    for (int i = lane; i < HP; i += 32) s += h[i] * W2[i];
    #pragma unroll
    for (int off = 16; off > 0; off >>= 1) s += __shfl_xor(s, off, 32);
    if (lane == 0) {
        float p = 1.f / (1.f + __expf(-(s + b2[0])));
        hb[t] = (p > 0.5f) ? 1 : 0;
    }
}

// ---------------------------------------------------------------------------
// Per-batch segment-id scan + patch counts (S = 1024 threads / block)
// ---------------------------------------------------------------------------
__global__ __launch_bounds__(1024)
void scan_kernel(const int* __restrict__ hb, int* __restrict__ seg,
                 float* __restrict__ counts, int* __restrict__ npatch, int S)
{
    __shared__ int sh[1024];
    __shared__ int sg[1024];
    __shared__ int st[1024];
    int b = blockIdx.x, t = threadIdx.x;
    int v = hb[b * S + t];
    sh[t] = v;
    __syncthreads();
    for (int off = 1; off < S; off <<= 1) {
        int x = sh[t];
        int y = (t >= off) ? sh[t - off] : 0;
        __syncthreads();
        sh[t] = x + y;
        __syncthreads();
    }
    int segid = sh[t] - v;
    sg[t] = segid;
    seg[b * S + t] = segid;
    counts[b * S + t] = 0.f;
    __syncthreads();
    if (t == 0 || sg[t - 1] != segid) st[segid] = t;
    __syncthreads();
    if (t == S - 1 || sg[t + 1] != segid)
        counts[b * S + segid] = (float)(t - st[segid] + 1);
    if (t == S - 1) npatch[b] = segid + 1;
}

// ---------------------------------------------------------------------------
// Segment mean: each thread owns one channel d, walks s sequentially.
// ---------------------------------------------------------------------------
__global__ __launch_bounds__(256)
void segmean_kernel(const float* __restrict__ X, const int* __restrict__ seg,
                    const float* __restrict__ counts, float* __restrict__ P,
                    int S, int D)
{
    int b = blockIdx.y;
    int d = blockIdx.x * 256 + threadIdx.x;
    const float* x   = X + (size_t)b * S * D + d;
    const int*   sgp = seg + b * S;
    const float* cnt = counts + b * S;
    float*       p   = P + (size_t)b * S * D + d;
    float acc = 0.f;
    int   cur = 0;
    for (int s = 0; s < S; s++) {
        int sid = sgp[s];
        if (sid != cur) {
            p[(size_t)cur * D] = acc / fmaxf(cnt[cur], 1.f);
            acc = 0.f;
            cur = sid;
        }
        acc += x[(size_t)s * D];
    }
    p[(size_t)cur * D] = acc / fmaxf(cnt[cur], 1.f);
}

// ---------------------------------------------------------------------------
// LayerNorm over D=1024 (no affine): Y = LN(X + R), R nullable.
// ---------------------------------------------------------------------------
__global__ __launch_bounds__(256)
void ln_kernel(const float* __restrict__ X, const float* __restrict__ R,
               float* __restrict__ Y, int D)
{
    int row = blockIdx.x, t = threadIdx.x;
    __shared__ float s1[256], s2[256];
    const float* x = X + (size_t)row * D;
    const float* r = R ? R + (size_t)row * D : nullptr;
    float vals[4];
    float a = 0.f, b = 0.f;
    #pragma unroll
    for (int j = 0; j < 4; j++) {
        int i = t + j * 256;
        float v = x[i] + (r ? r[i] : 0.f);
        vals[j] = v;
        a += v;
        b += v * v;
    }
    s1[t] = a; s2[t] = b;
    __syncthreads();
    for (int off = 128; off > 0; off >>= 1) {
        if (t < off) { s1[t] += s1[t + off]; s2[t] += s2[t + off]; }
        __syncthreads();
    }
    float mean = s1[0] / (float)D;
    float var  = s2[0] / (float)D - mean * mean;
    float inv  = rsqrtf(var + 1e-5f);
    float* y = Y + (size_t)row * D;
    #pragma unroll
    for (int j = 0; j < 4; j++) {
        int i = t + j * 256;
        y[i] = (vals[j] - mean) * inv;
    }
}

// ---------------------------------------------------------------------------
// Masked, scaled softmax in-place over rows of scores[z][row][0..N).
// ---------------------------------------------------------------------------
__global__ __launch_bounds__(256)
void softmax_kernel(float* __restrict__ S, const int* __restrict__ npatch,
                    int Mrows, int N, int nh, float scale, int masked)
{
    int z = blockIdx.y, row = blockIdx.x, t = threadIdx.x;
    float* p = S + ((size_t)z * Mrows + row) * N;
    int limit = masked ? npatch[z / nh] : N;
    __shared__ float red[256];
    float mx = -3.0e38f;
    for (int i = t; i < N; i += 256)
        if (i < limit) mx = fmaxf(mx, p[i] * scale);
    red[t] = mx;
    __syncthreads();
    for (int off = 128; off > 0; off >>= 1) {
        if (t < off) red[t] = fmaxf(red[t], red[t + off]);
        __syncthreads();
    }
    mx = red[0];
    __syncthreads();
    float sum = 0.f;
    for (int i = t; i < N; i += 256) {
        float e = (i < limit) ? __expf(p[i] * scale - mx) : 0.f;
        p[i] = e;
        sum += e;
    }
    red[t] = sum;
    __syncthreads();
    for (int off = 128; off > 0; off >>= 1) {
        if (t < off) red[t] += red[t + off];
        __syncthreads();
    }
    float inv = 1.f / red[0];
    for (int i = t; i < N; i += 256) p[i] *= inv;
}

// ---------------------------------------------------------------------------
// Embedding gather: out[row] = emb[idx[row]]  (idx is int64)
// ---------------------------------------------------------------------------
__global__ __launch_bounds__(256)
void gather_kernel(const long long* __restrict__ idx, const float* __restrict__ emb,
                   float* __restrict__ out, int D)
{
    int row = blockIdx.x;
    long long id = idx[row];
    const float* e = emb + (size_t)id * D;
    float* o = out + (size_t)row * D;
    for (int i = threadIdx.x; i < D; i += blockDim.x) o[i] = e[i];
}

// ---------------------------------------------------------------------------
// Host orchestration
// ---------------------------------------------------------------------------
extern "C" void kernel_launch(void* const* d_in, const int* in_sizes, int n_in,
                              void* d_out, int out_size, void* d_ws, size_t ws_size,
                              hipStream_t stream)
{
    (void)in_sizes; (void)n_in; (void)out_size; (void)ws_size;

    const int Bb = 2, S = 1024, Dd = 1024, HP = 512, FFd = 4096;
    const int NHh = 16, HDd = 64, Vv = 256;
    const int T = Bb * S;
    const size_t DDs = (size_t)Dd * Dd;

    const float*     latent     = (const float*)d_in[0];
    const float*     pW1        = (const float*)d_in[1];
    const float*     pb1        = (const float*)d_in[2];
    const float*     pW2        = (const float*)d_in[3];
    const float*     pb2        = (const float*)d_in[4];
    const float*     ltWqkv     = (const float*)d_in[5];
    const float*     ltWo       = (const float*)d_in[6];
    const float*     ltff1      = (const float*)d_in[7];
    const float*     ltff2      = (const float*)d_in[8];
    const float*     emb        = (const float*)d_in[9];
    const float*     decWq      = (const float*)d_in[10];
    const float*     decWk      = (const float*)d_in[11];
    const float*     decWv      = (const float*)d_in[12];
    const float*     decMhaWqkv = (const float*)d_in[13];
    const float*     decMhaWo   = (const float*)d_in[14];
    const float*     decDense   = (const float*)d_in[15];
    const float*     decSaWqkv  = (const float*)d_in[16];
    const float*     decSaWo    = (const float*)d_in[17];
    const float*     decff1     = (const float*)d_in[18];
    const float*     decff2     = (const float*)d_in[19];
    const float*     finalW     = (const float*)d_in[20];
    const long long* byteSeq    = (const long long*)d_in[21];

    // ---- workspace carve-up ----
    char* wsp = (char*)d_ws;
    auto carve = [&](size_t bytes) -> char* {
        char* p = wsp;
        wsp += (bytes + 255) & ~(size_t)255;
        return p;
    };
    float* H      = (float*)carve((size_t)T * HP * 4);
    int*   hb     = (int*)  carve((size_t)T * 4);
    int*   seg    = (int*)  carve((size_t)T * 4);
    float* counts = (float*)carve((size_t)T * 4);
    int*   npatch = (int*)  carve(256);
    float* patch  = (float*)carve((size_t)T * Dd * 4);
    float* X      = (float*)carve((size_t)T * Dd * 4);
    float* XO     = (float*)carve((size_t)T * Dd * 4);
    float* A1     = (float*)carve((size_t)T * Dd * 4);
    float* A2     = (float*)carve((size_t)T * Dd * 4);
    float* A3     = (float*)carve((size_t)T * Dd * 4);
    float* A4     = (float*)carve((size_t)T * Dd * 4);
    float* A5     = (float*)carve((size_t)T * Dd * 4);
    float* A6     = (float*)carve((size_t)T * Dd * 4);
    float* BR     = (float*)carve((size_t)T * Dd * 4);
    float* CR     = (float*)carve((size_t)T * Dd * 4);
    float* ffbuf  = (float*)carve((size_t)T * FFd * 4);
    float* scores = (float*)carve((size_t)Bb * NHh * S * S * 4);
    // bf16 pre-transposed weights [N,K]
    bf16* pW1t        = (bf16*)carve((size_t)HP * Dd * 2);
    bf16* ltWqkvT     = (bf16*)carve(3 * DDs * 2);
    bf16* ltWoT       = (bf16*)carve(DDs * 2);
    bf16* ltff1T      = (bf16*)carve((size_t)FFd * Dd * 2);
    bf16* ltff2T      = (bf16*)carve((size_t)Dd * FFd * 2);
    bf16* decWqT      = (bf16*)carve(2 * DDs * 2);
    bf16* decWkT      = (bf16*)carve(2 * DDs * 2);
    bf16* decWvT      = (bf16*)carve(2 * DDs * 2);
    bf16* decMhaWqkvT = (bf16*)carve(6 * DDs * 2);
    bf16* decMhaWoT   = (bf16*)carve(2 * DDs * 2);
    bf16* decDenseT   = (bf16*)carve(2 * DDs * 2);
    bf16* decSaWqkvT  = (bf16*)carve(6 * DDs * 2);
    bf16* decSaWoT    = (bf16*)carve(2 * DDs * 2);
    bf16* decff1T     = (bf16*)carve(2 * (size_t)FFd * Dd * 2);
    bf16* decff2T     = (bf16*)carve(2 * (size_t)Dd * FFd * 2);
    bf16* finalWT     = (bf16*)carve((size_t)Vv * Dd * 2);

    // ---- launch helpers ----
    auto conv = [&](const float* src, bf16* dst, int K, int N) {
        convtr_kernel<<<dim3((N + 31) / 32, (K + 31) / 32), 256, 0, stream>>>(src, dst, K, N);
    };
    auto gemmW = [&](const float* A, const bf16* Wt, float* C, int M, int N, int K,
                     const float* bias, int relu, const float* resid) {
        dim3 grid(N / TBN, M / TBM, 1);
        gemm_wmma_wbf16_kernel<<<grid, 256, 0, stream>>>(A, Wt, C, bias, resid, M, N, K, relu);
    };
    auto gemm_qk = [&](const float* Q, const float* Kp, float* Sc) {
        dim3 grid((S + TBN - 1) / TBN, (S + TBM - 1) / TBM, Bb * NHh);
        gemm_wmma_kernel<<<grid, 256, 0, stream>>>(Q, Kp, Sc,
            S, S, HDd, Dd, Dd, S, /*transB=*/1,
            (long long)S * Dd, (long long)HDd,
            (long long)S * Dd, (long long)HDd,
            (long long)NHh * S * S, (long long)S * S, NHh);
    };
    auto gemm_pv = [&](const float* P, const float* Vp, float* Ctx) {
        dim3 grid((HDd + TBN - 1) / TBN, (S + TBM - 1) / TBM, Bb * NHh);
        gemm_wmma_kernel<<<grid, 256, 0, stream>>>(P, Vp, Ctx,
            S, HDd, S, S, Dd, Dd, /*transB=*/0,
            (long long)NHh * S * S, (long long)S * S,
            (long long)S * Dd, (long long)HDd,
            (long long)S * Dd, (long long)HDd, NHh);
    };
    auto ln = [&](const float* Xi, const float* R, float* Y) {
        ln_kernel<<<dim3(T), 256, 0, stream>>>(Xi, R, Y, Dd);
    };
    auto smax = [&](float* Sc, int masked) {
        softmax_kernel<<<dim3(S, Bb * NHh), 256, 0, stream>>>(
            Sc, npatch, S, S, NHh, 0.125f, masked);
    };

    // ================= Weight pre-transpose/convert (fp32 -> bf16 [N,K]) ====
    conv(pW1, pW1t, Dd, HP);
    for (int j = 0; j < 3; j++) conv(ltWqkv + j * DDs, ltWqkvT + j * DDs, Dd, Dd);
    conv(ltWo, ltWoT, Dd, Dd);
    conv(ltff1, ltff1T, Dd, FFd);
    conv(ltff2, ltff2T, FFd, Dd);
    for (int i = 0; i < 2; i++) {
        conv(decWq + i * DDs, decWqT + i * DDs, Dd, Dd);
        conv(decWk + i * DDs, decWkT + i * DDs, Dd, Dd);
        conv(decWv + i * DDs, decWvT + i * DDs, Dd, Dd);
        conv(decMhaWo + i * DDs, decMhaWoT + i * DDs, Dd, Dd);
        conv(decDense + i * DDs, decDenseT + i * DDs, Dd, Dd);
        conv(decSaWo + i * DDs, decSaWoT + i * DDs, Dd, Dd);
        conv(decff1 + (size_t)i * Dd * FFd, decff1T + (size_t)i * FFd * Dd, Dd, FFd);
        conv(decff2 + (size_t)i * FFd * Dd, decff2T + (size_t)i * Dd * FFd, FFd, Dd);
    }
    for (int j = 0; j < 6; j++) {
        conv(decMhaWqkv + j * DDs, decMhaWqkvT + j * DDs, Dd, Dd);
        conv(decSaWqkv  + j * DDs, decSaWqkvT  + j * DDs, Dd, Dd);
    }
    conv(finalW, finalWT, Dd, Vv);

    // ================= Stage A: binary patching =================
    gemmW(latent, pW1t, H, T, HP, Dd, pb1, 1, nullptr);
    boundary_kernel<<<dim3(T / 8), 256, 0, stream>>>(H, pW2, pb2, hb, HP);
    scan_kernel<<<dim3(Bb), 1024, 0, stream>>>(hb, seg, counts, npatch, S);
    hipMemsetAsync(patch, 0, (size_t)T * Dd * 4, stream);
    segmean_kernel<<<dim3(Dd / 256, Bb), 256, 0, stream>>>(latent, seg, counts, patch, S, Dd);

    // ================= Stage B: latent transformer (post-norm) =================
    gemmW(patch, ltWqkvT + 0 * DDs, A1, T, Dd, Dd, nullptr, 0, nullptr); // Q
    gemmW(patch, ltWqkvT + 1 * DDs, A2, T, Dd, Dd, nullptr, 0, nullptr); // K
    gemmW(patch, ltWqkvT + 2 * DDs, A3, T, Dd, Dd, nullptr, 0, nullptr); // V
    gemm_qk(A1, A2, scores);
    smax(scores, 1);
    gemm_pv(scores, A3, A4);
    gemmW(A4, ltWoT, A5, T, Dd, Dd, nullptr, 0, nullptr);
    ln(A5, patch, X);                                   // x = LN(patch + attn)
    gemmW(X, ltff1T, ffbuf, T, FFd, Dd, nullptr, 1, nullptr);
    gemmW(ffbuf, ltff2T, A1, T, Dd, FFd, nullptr, 0, nullptr);
    ln(A1, X, XO);                                      // x = LN(x + ffn)

    // ================= Stage C: local decoder =================
    gather_kernel<<<dim3(T), 256, 0, stream>>>(byteSeq, emb, BR, Dd);
    for (int i = 0; i < 2; i++) {
        const bf16* WqT  = decWqT + (size_t)i * DDs;
        const bf16* WkT  = decWkT + (size_t)i * DDs;
        const bf16* WvT  = decWvT + (size_t)i * DDs;
        const bf16* mW0T = decMhaWqkvT + (size_t)(i * 3 + 0) * DDs;
        const bf16* mW1T = decMhaWqkvT + (size_t)(i * 3 + 1) * DDs;
        const bf16* mW2T = decMhaWqkvT + (size_t)(i * 3 + 2) * DDs;
        const bf16* mWoT = decMhaWoT + (size_t)i * DDs;
        const bf16* dnsT = decDenseT + (size_t)i * DDs;
        const bf16* sW0T = decSaWqkvT + (size_t)(i * 3 + 0) * DDs;
        const bf16* sW1T = decSaWqkvT + (size_t)(i * 3 + 1) * DDs;
        const bf16* sW2T = decSaWqkvT + (size_t)(i * 3 + 2) * DDs;
        const bf16* sWoT = decSaWoT + (size_t)i * DDs;
        const bf16* ff1T = decff1T + (size_t)i * FFd * Dd;
        const bf16* ff2T = decff2T + (size_t)i * Dd * FFd;

        gemmW(BR, WqT, A1, T, Dd, Dd, nullptr, 0, nullptr); ln(A1, nullptr, A2);
        gemmW(XO, WkT, A1, T, Dd, Dd, nullptr, 0, nullptr); ln(A1, nullptr, A3);
        gemmW(XO, WvT, A1, T, Dd, Dd, nullptr, 0, nullptr); ln(A1, nullptr, A4);
        gemmW(A2, mW0T, A5, T, Dd, Dd, nullptr, 0, nullptr); // qh
        gemmW(A3, mW1T, A6, T, Dd, Dd, nullptr, 0, nullptr); // kh
        gemmW(A4, mW2T, A1, T, Dd, Dd, nullptr, 0, nullptr); // vh
        gemm_qk(A5, A6, scores);
        smax(scores, 1);
        gemm_pv(scores, A1, A2);
        gemmW(A2, mWoT, A3, T, Dd, Dd, nullptr, 0, nullptr);
        gemmW(A3, dnsT, CR, T, Dd, Dd, nullptr, 0, BR);      // cr = a@dense + byte_repr
        gemmW(CR, sW0T, A4, T, Dd, Dd, nullptr, 0, nullptr);
        gemmW(CR, sW1T, A5, T, Dd, Dd, nullptr, 0, nullptr);
        gemmW(CR, sW2T, A6, T, Dd, Dd, nullptr, 0, nullptr);
        gemm_qk(A4, A5, scores);
        smax(scores, 0);
        gemm_pv(scores, A6, A1);
        gemmW(A1, sWoT, A2, T, Dd, Dd, nullptr, 0, nullptr);
        ln(A2, CR, A3);                                      // y = LN(cr + s)
        gemmW(A3, ff1T, ffbuf, T, FFd, Dd, nullptr, 1, nullptr);
        gemmW(ffbuf, ff2T, A4, T, Dd, FFd, nullptr, 0, nullptr);
        ln(A4, A3, BR);                                      // byte_repr
    }

    // ================= Final projection to byte logits =================
    gemmW(BR, finalWT, (float*)d_out, T, Vv, Dd, nullptr, 0, nullptr);
}
